// GraphSAGE_5119601017051
// MI455X (gfx1250) — matrix operand order repaired and checked by reading, not compiled, not run on hardware
//
#include <hip/hip_runtime.h>
#include <hip/hip_bf16.h>

typedef __bf16 bf16_t;
typedef __attribute__((ext_vector_type(16))) __bf16 v16bf;
typedef __attribute__((ext_vector_type(8)))  __bf16 v8bf;
typedef __attribute__((ext_vector_type(8)))  float  v8f;

#define HIDF 128
#define LN_EPS 1e-5f

// ---------------- utility: zero fp32 buffer ----------------
__global__ void zero_f32_kernel(float* __restrict__ p, int n) {
    int i = blockIdx.x * blockDim.x + threadIdx.x;
    if (i < n) p[i] = 0.0f;
}

// ---------------- weight convert: W[K=128][N=128] f32 -> Wt[N=128][K=128] bf16
__global__ void convert_weight_kernel(const float* __restrict__ W, bf16_t* __restrict__ Wt) {
    int i = blockIdx.x * blockDim.x + threadIdx.x;   // 0..16383
    int k = i >> 7;
    int n = i & 127;
    Wt[n * 128 + k] = (bf16_t)W[i];
}

// ---------------- edge scatter: one wave32 per edge ----------------
__global__ void scatter_kernel(const float* __restrict__ feat,
                               const int* __restrict__ src,
                               const int* __restrict__ dst,
                               float* __restrict__ msg,
                               float* __restrict__ deg,
                               int E) {
    int gid  = blockIdx.x * blockDim.x + threadIdx.x;
    int e    = gid >> 5;
    int lane = gid & 31;
    if (e >= E) return;
    int s = src[e];
    int d = dst[e];
    const float* fs = feat + (long)s * HIDF;
    float*       md = msg  + (long)d * HIDF;
#pragma unroll
    for (int j = 0; j < 4; ++j) {
        int f = lane + 32 * j;
        atomicAdd(&md[f], fs[f]);
    }
    if (lane == 0) atomicAdd(&deg[d], 1.0f);
}

// ---------------- fused SAGEConv + bias + LayerNorm + ReLU ----------------
// 256 threads (8 waves) per 16-node tile. Wave w computes output columns
// [16w, 16w+16) of  mean@Wl + x@Wr  with v_wmma_f32_16x16x32_bf16.
__global__ __launch_bounds__(256) void sage_block_kernel(
    const float* __restrict__ xin,    // [n,128] root features (fp32)
    const float* __restrict__ msg,    // [n,128] neighbor sums (fp32)
    const float* __restrict__ deg,    // [n]
    const bf16_t* __restrict__ Wl_t,  // [128 n][128 k] bf16 (transposed)
    const bf16_t* __restrict__ Wr_t,  // [128 n][128 k] bf16
    const float* __restrict__ bias,   // [128]
    const float* __restrict__ lnw,    // [128]
    const float* __restrict__ lnb,    // [128]
    float* __restrict__ hout,         // [n,128]
    int n) {
    __shared__ __align__(16) bf16_t sAm[16 * 128];  // mean-agg tile, bf16
    __shared__ __align__(16) bf16_t sAx[16 * 128];  // root tile, bf16
    __shared__ __align__(16) float  sD[16 * 128];   // pre-LN f32 tile

    const int t     = threadIdx.x;
    const int node0 = blockIdx.x * 16;

    // ---- stage A tiles: each thread converts 8 contiguous elements ----
    {
        int row = t >> 4;                  // 0..15
        int cb  = (t & 15) * 8;            // 0,8,...,120
        int nd  = node0 + row;
        if (nd > n - 1) nd = n - 1;        // clamp (n is a multiple of 16 here)
        long base = (long)nd * HIDF + cb;
        float inv = 1.0f / fmaxf(deg[nd], 1.0f);
        float4 m0 = *(const float4*)(msg + base);
        float4 m1 = *(const float4*)(msg + base + 4);
        float4 x0 = *(const float4*)(xin + base);
        float4 x1 = *(const float4*)(xin + base + 4);
        v8bf vm, vx;
        vm[0] = (bf16_t)(m0.x * inv); vm[1] = (bf16_t)(m0.y * inv);
        vm[2] = (bf16_t)(m0.z * inv); vm[3] = (bf16_t)(m0.w * inv);
        vm[4] = (bf16_t)(m1.x * inv); vm[5] = (bf16_t)(m1.y * inv);
        vm[6] = (bf16_t)(m1.z * inv); vm[7] = (bf16_t)(m1.w * inv);
        vx[0] = (bf16_t)x0.x; vx[1] = (bf16_t)x0.y;
        vx[2] = (bf16_t)x0.z; vx[3] = (bf16_t)x0.w;
        vx[4] = (bf16_t)x1.x; vx[5] = (bf16_t)x1.y;
        vx[6] = (bf16_t)x1.z; vx[7] = (bf16_t)x1.w;
        *(v8bf*)&sAm[row * 128 + cb] = vm;
        *(v8bf*)&sAx[row * 128 + cb] = vx;
    }
    __syncthreads();

    const int wave = t >> 5;
    const int lane = t & 31;
    const int nloc = lane & 15;            // A row / B column within tile
    const int ncol = wave * 16 + nloc;     // output feature column

    // Fragment K offsets per ISA VGPR layouts (wave32, 16-bit data):
    //   A 16x32: lane<16 -> K {0..7, 16..23};  lane>=16 -> K {8..15, 24..31}
    //   B 32x16: lane<16 -> K {0..7, 16..23};  lane>=16 -> K {8..15, 24..31}
    const int k0a = (lane < 16) ? 0 : 8;
    const int k0b = (lane < 16) ? 0 : 8;

    v8f acc = {};
#pragma unroll
    for (int kb = 0; kb < 4; ++kb) {
        const int ka = kb * 32 + k0a;
        const int kc = kb * 32 + k0b;
        v16bf am, ax, bl, br;
        {
            uint4* p = (uint4*)&am;
            p[0] = *(const uint4*)&sAm[nloc * 128 + ka];
            p[1] = *(const uint4*)&sAm[nloc * 128 + ka + 16];
        }
        {
            uint4* p = (uint4*)&ax;
            p[0] = *(const uint4*)&sAx[nloc * 128 + ka];
            p[1] = *(const uint4*)&sAx[nloc * 128 + ka + 16];
        }
        {
            uint4* p = (uint4*)&bl;
            p[0] = *(const uint4*)&Wl_t[(long)ncol * 128 + kc];
            p[1] = *(const uint4*)&Wl_t[(long)ncol * 128 + kc + 16];
        }
        {
            uint4* p = (uint4*)&br;
            p[0] = *(const uint4*)&Wr_t[(long)ncol * 128 + kc];
            p[1] = *(const uint4*)&Wr_t[(long)ncol * 128 + kc + 16];
        }
        acc = __builtin_amdgcn_wmma_f32_16x16x32_bf16(false, am, false, bl,
                                                      (short)0, acc, false, false);
        acc = __builtin_amdgcn_wmma_f32_16x16x32_bf16(false, ax, false, br,
                                                      (short)0, acc, false, false);
    }

    // ---- spill D tile (+bias) to LDS. C/D layout: VGPR r -> row r (+8 for hi lanes)
    {
        float bc = bias[ncol];
        int rbase = (lane < 16) ? 0 : 8;
#pragma unroll
        for (int r = 0; r < 8; ++r)
            sD[(rbase + r) * 128 + ncol] = acc[r] + bc;
    }
    __syncthreads();

    // ---- LayerNorm + ReLU: wave w handles rows 2w and 2w+1 ----
#pragma unroll
    for (int rr = 0; rr < 2; ++rr) {
        int row = wave * 2 + rr;
        float v0 = sD[row * 128 + lane];
        float v1 = sD[row * 128 + lane + 32];
        float v2 = sD[row * 128 + lane + 64];
        float v3 = sD[row * 128 + lane + 96];
        float s = v0 + v1 + v2 + v3;
#pragma unroll
        for (int off = 16; off > 0; off >>= 1) s += __shfl_xor(s, off);
        float mu = s * (1.0f / 128.0f);
        float d0 = v0 - mu, d1 = v1 - mu, d2 = v2 - mu, d3 = v3 - mu;
        float q = d0 * d0 + d1 * d1 + d2 * d2 + d3 * d3;
#pragma unroll
        for (int off = 16; off > 0; off >>= 1) q += __shfl_xor(q, off);
        float rs = rsqrtf(q * (1.0f / 128.0f) + LN_EPS);
        int nd = node0 + row;
        if (nd < n) {
            long ob = (long)nd * HIDF;
            hout[ob + lane]      = fmaxf(d0 * rs * lnw[lane]      + lnb[lane],      0.0f);
            hout[ob + lane + 32] = fmaxf(d1 * rs * lnw[lane + 32] + lnb[lane + 32], 0.0f);
            hout[ob + lane + 64] = fmaxf(d2 * rs * lnw[lane + 64] + lnb[lane + 64], 0.0f);
            hout[ob + lane + 96] = fmaxf(d3 * rs * lnw[lane + 96] + lnb[lane + 96], 0.0f);
        }
    }
}

// ---------------- head: out[node] = h[node,:] . fcW + fcb (wave per node)
__global__ void head_kernel(const float* __restrict__ h,
                            const float* __restrict__ fcW,
                            const float* __restrict__ fcb,
                            float* __restrict__ out, int n) {
    int gid  = blockIdx.x * blockDim.x + threadIdx.x;
    int node = gid >> 5;
    int lane = gid & 31;
    if (node >= n) return;
    const float* hp = h + (long)node * HIDF;
    float a = 0.0f;
#pragma unroll
    for (int j = 0; j < 4; ++j) a += hp[lane + 32 * j] * fcW[lane + 32 * j];
#pragma unroll
    for (int off = 16; off > 0; off >>= 1) a += __shfl_xor(a, off);
    if (lane == 0) out[node] = a + fcb[0];
}

extern "C" void kernel_launch(void* const* d_in, const int* in_sizes, int n_in,
                              void* d_out, int out_size, void* d_ws, size_t ws_size,
                              hipStream_t stream) {
    const float* x    = (const float*)d_in[0];
    const int*   ei   = (const int*)d_in[1];
    const float* W1l  = (const float*)d_in[2];
    const float* W1r  = (const float*)d_in[3];
    const float* b1   = (const float*)d_in[4];
    const float* W2l  = (const float*)d_in[5];
    const float* W2r  = (const float*)d_in[6];
    const float* b2   = (const float*)d_in[7];
    const float* ln1w = (const float*)d_in[8];
    const float* ln1b = (const float*)d_in[9];
    const float* ln2w = (const float*)d_in[10];
    const float* ln2b = (const float*)d_in[11];
    const float* fcW  = (const float*)d_in[12];
    const float* fcb  = (const float*)d_in[13];
    float* out = (float*)d_out;

    const int n = in_sizes[0] / HIDF;   // 100000
    const int E = in_sizes[1] / 2;      // 1600000

    // workspace carve-up
    float*  msg  = (float*)d_ws;                         // n*128
    float*  deg  = msg + (size_t)n * HIDF;               // n
    float*  h1   = deg + n;                              // n*128
    float*  h2   = h1 + (size_t)n * HIDF;                // n*128
    bf16_t* wt   = (bf16_t*)(h2 + (size_t)n * HIDF);     // 4 * 16384 bf16
    bf16_t* w1lt = wt;
    bf16_t* w1rt = wt + 16384;
    bf16_t* w2lt = wt + 32768;
    bf16_t* w2rt = wt + 49152;

    // 0) convert all four weight matrices to transposed bf16
    convert_weight_kernel<<<64, 256, 0, stream>>>(W1l, w1lt);
    convert_weight_kernel<<<64, 256, 0, stream>>>(W1r, w1rt);
    convert_weight_kernel<<<64, 256, 0, stream>>>(W2l, w2lt);
    convert_weight_kernel<<<64, 256, 0, stream>>>(W2r, w2rt);

    const int zn = n * HIDF + n;        // msg + deg are contiguous
    const int zb = (zn + 255) / 256;
    const int sb = (E * 32 + 255) / 256;
    const int gb = (n + 15) / 16;
    const int hb = (n * 32 + 255) / 256;

    // layer 1
    zero_f32_kernel<<<zb, 256, 0, stream>>>(msg, zn);
    scatter_kernel<<<sb, 256, 0, stream>>>(x, ei, ei + E, msg, deg, E);
    sage_block_kernel<<<gb, 256, 0, stream>>>(x, msg, deg, w1lt, w1rt, b1,
                                              ln1w, ln1b, h1, n);
    // layer 2
    zero_f32_kernel<<<zb, 256, 0, stream>>>(msg, zn);
    scatter_kernel<<<sb, 256, 0, stream>>>(h1, ei, ei + E, msg, deg, E);
    sage_block_kernel<<<gb, 256, 0, stream>>>(h1, msg, deg, w2lt, w2rt, b2,
                                              ln2w, ln2b, h2, n);
    // head
    head_kernel<<<hb, 256, 0, stream>>>(h2, fcW, fcb, out, n);
}
